// DecoupledAttention_40742059770085
// MI455X (gfx1250) — compile-verified, matrix operand order checked by reading, not run on hardware
//
#include <hip/hip_runtime.h>

typedef __attribute__((ext_vector_type(8)))  __bf16 v8bf;
typedef __attribute__((ext_vector_type(16))) __bf16 v16bf;
typedef __attribute__((ext_vector_type(8)))  float  v8f;

#define INV_SQRT_N 0.03125f       // 1/sqrt(1024)
#define SCALING_C  0.015625f      // 1/64
#define ALPHA_C    0.00902110935f // sqrt(1/(12*1024))

// ---------------------------------------------------------------------------
// WMMA 16-bit fragment loader (A: 16xK rows, B: Kx16 cols -- same lane layout).
// Lane l (0..31): r = l&15, sel = l>>4; holds K chunks [k0+sel*8 .. +7] and
// [k0+16+sel*8 .. +7]  (per CDNA5 ISA 16-bit A-matrix 16x32 layout).
// ---------------------------------------------------------------------------
__device__ __forceinline__ v16bf load_frag16(const __bf16* __restrict__ base,
                                             int ld, int k0, int lane) {
  const int r   = lane & 15;
  const int sel = lane >> 4;
  const __bf16* p = base + (size_t)r * ld + k0 + sel * 8;
  union { v16bf v; struct { v8bf lo, hi; } s; } u;
  u.s.lo = *(const v8bf*)(p);
  u.s.hi = *(const v8bf*)(p + 16);
  return u.v;
}

__device__ __forceinline__ v8f wmma_bf16(v16bf a, v16bf b, v8f c) {
  return __builtin_amdgcn_wmma_f32_16x16x32_bf16(false, a, false, b,
                                                 (short)0, c, false, false);
}

__device__ __forceinline__ void nt_store(float* p, float v) {
  __builtin_nontemporal_store(v, p);
}
__device__ __forceinline__ float nt_load(const float* p) {
  return __builtin_nontemporal_load(p);
}

// ---------------------------------------------------------------------------
// fp32 -> bf16 convert
// ---------------------------------------------------------------------------
__global__ void cvt_f32_bf16(const float* __restrict__ in,
                             __bf16* __restrict__ out, int n) {
  int i = blockIdx.x * blockDim.x + threadIdx.x;
  if (i < n) out[i] = (__bf16)in[i];
}

// ---------------------------------------------------------------------------
// Fused projection pair: P0 = s*(A @ W0^T), P1 = s*(A @ W1^T), diff = P0-P1.
// A: [4096][1024] bf16, W*: [1024][1024] bf16 (row-major -> K-contig).
// One wave computes a 32(M)x16(N) tile for BOTH weights: 4 frag loads,
// 4 WMMAs per k-step.  Waves: (4096/32)*(1024/16) = 8192 -> 1024 blocks.
// Outputs in head layout [b*16+h][s][d]; optional transposed [bh][d][s].
// ---------------------------------------------------------------------------
__global__ void proj_pair_kernel(const __bf16* __restrict__ A,
                                 const __bf16* __restrict__ W0,
                                 const __bf16* __restrict__ W1,
                                 float*  __restrict__ diff_out,  // [b,h,s,d]
                                 __bf16* __restrict__ P0,        // [bh][s][d] or null
                                 __bf16* __restrict__ P1,        // [bh][s][d] or null
                                 __bf16* __restrict__ P0t) {     // [bh][d][s] or null
  const int wave = (blockIdx.x * blockDim.x + threadIdx.x) >> 5;
  const int lane = threadIdx.x & 31;
  const int mt = wave >> 6;   // 0..127  (32-row tiles)
  const int nt = wave & 63;   // 0..63
  const int m0 = mt * 32, n0 = nt * 16;

  const __bf16* A0b = A  + (size_t)m0 * 1024;
  const __bf16* A1b = A  + (size_t)(m0 + 16) * 1024;
  const __bf16* W0b = W0 + (size_t)n0 * 1024;
  const __bf16* W1b = W1 + (size_t)n0 * 1024;

  v8f acc00 = {}, acc01 = {}, acc10 = {}, acc11 = {};
  for (int k0 = 0; k0 < 1024; k0 += 32) {
    v16bf a0 = load_frag16(A0b, 1024, k0, lane);
    v16bf a1 = load_frag16(A1b, 1024, k0, lane);
    v16bf b0 = load_frag16(W0b, 1024, k0, lane);
    v16bf b1 = load_frag16(W1b, 1024, k0, lane);
    acc00 = wmma_bf16(a0, b0, acc00);   // W0, rows m0..m0+15
    acc01 = wmma_bf16(a1, b0, acc01);   // W0, rows m0+16..m0+31
    acc10 = wmma_bf16(a0, b1, acc10);   // W1, rows m0..m0+15
    acc11 = wmma_bf16(a1, b1, acc11);   // W1, rows m0+16..m0+31
  }

  const int sel = lane >> 4;
  const int n = n0 + (lane & 15);
  const int hd = n >> 6, d = n & 63;
#pragma unroll
  for (int half = 0; half < 2; ++half) {
    const v8f& a0 = half ? acc01 : acc00;
    const v8f& a1 = half ? acc11 : acc10;
#pragma unroll
    for (int v = 0; v < 8; ++v) {
      const int m = m0 + half * 16 + sel * 8 + v;
      const int b = m >> 10, s = m & 1023;
      const int bh = b * 16 + hd;
      const float p0 = INV_SQRT_N * a0[v];
      const float p1 = INV_SQRT_N * a1[v];
      const size_t hi = ((size_t)bh * 1024 + s) * 64 + d;
      nt_store(diff_out + hi, p0 - p1);
      if (P0)  P0[hi] = (__bf16)p0;
      if (P1)  P1[hi] = (__bf16)p1;
      if (P0t) P0t[((size_t)bh * 64 + d) * 1024 + s] = (__bf16)p0;
    }
  }
}

// ---------------------------------------------------------------------------
// Energy: e = Q@K^T, er = Qr@Kr^T per (bh).  K-dim = 64 (2 WMMA steps).
// One wave computes a 16(q)x64(k) strip for BOTH energies: Q/Qr fragments
// loaded once per 4 k-column tiles; 10 loads, 8 WMMAs per k-step.
// Waves: 64(bh) * 64(qt) * 16(kgroup) = 65536 -> 8192 blocks.
// ---------------------------------------------------------------------------
__global__ void energy_kernel(const __bf16* __restrict__ Q,
                              const __bf16* __restrict__ K,
                              const __bf16* __restrict__ Qr,
                              const __bf16* __restrict__ Kr,
                              float*  __restrict__ ediff_out,   // [bh][q][k]
                              __bf16* __restrict__ ebf) {       // [bh][q][k]
  const int wave = (blockIdx.x * blockDim.x + threadIdx.x) >> 5;
  const int lane = threadIdx.x & 31;
  const int kg = wave & 15;            // 64-wide k group
  const int qt = (wave >> 4) & 63;
  const int bh = wave >> 10;

  const size_t qoff = ((size_t)bh * 1024 + qt * 16) * 64;
  const size_t koff = ((size_t)bh * 1024 + kg * 64) * 64;
  const __bf16* Qb  = Q  + qoff;
  const __bf16* Qrb = Qr + qoff;
  const __bf16* Kb  = K  + koff;
  const __bf16* Krb = Kr + koff;

  v8f e[4]  = {{}, {}, {}, {}};
  v8f er[4] = {{}, {}, {}, {}};
#pragma unroll
  for (int k0 = 0; k0 < 64; k0 += 32) {
    v16bf aq = load_frag16(Qb,  64, k0, lane);
    v16bf ar = load_frag16(Qrb, 64, k0, lane);
#pragma unroll
    for (int j = 0; j < 4; ++j) {
      v16bf bk = load_frag16(Kb  + (size_t)(j * 16) * 64, 64, k0, lane);
      e[j] = wmma_bf16(aq, bk, e[j]);
      v16bf br = load_frag16(Krb + (size_t)(j * 16) * 64, 64, k0, lane);
      er[j] = wmma_bf16(ar, br, er[j]);
    }
  }

  const int sel = lane >> 4;
#pragma unroll
  for (int j = 0; j < 4; ++j) {
    const int n = kg * 64 + j * 16 + (lane & 15);
#pragma unroll
    for (int v = 0; v < 8; ++v) {
      const int m = qt * 16 + sel * 8 + v;
      const size_t idx = ((size_t)bh * 1024 + m) * 1024 + n;
      nt_store(ediff_out + idx, SCALING_C * (e[j][v] - er[j][v]));
      ebf[idx] = (__bf16)(SCALING_C * e[j][v]);
    }
  }
}

// ---------------------------------------------------------------------------
// Row softmax over 1024 fp32, NT in / NT out (stream-once data).
// One block (256 thr) per row.
// ---------------------------------------------------------------------------
__global__ void softmax1024_f32(const float* __restrict__ in,
                                float* __restrict__ out) {
  __shared__ float red[256];
  const size_t row = blockIdx.x;
  const float* rp = in + row * 1024;
  float* wp = out + row * 1024;
  const int tid = threadIdx.x;
  float v[4]; float mx = -3.4e38f;
#pragma unroll
  for (int i = 0; i < 4; ++i) { v[i] = nt_load(rp + tid + 256 * i); mx = fmaxf(mx, v[i]); }
  red[tid] = mx; __syncthreads();
  for (int s = 128; s > 0; s >>= 1) {
    if (tid < s) red[tid] = fmaxf(red[tid], red[tid + s]);
    __syncthreads();
  }
  mx = red[0]; __syncthreads();
  float sum = 0.f;
#pragma unroll
  for (int i = 0; i < 4; ++i) { v[i] = __expf(v[i] - mx); sum += v[i]; }
  red[tid] = sum; __syncthreads();
  for (int s = 128; s > 0; s >>= 1) {
    if (tid < s) red[tid] += red[tid + s];
    __syncthreads();
  }
  const float inv = 1.0f / red[0];
#pragma unroll
  for (int i = 0; i < 4; ++i) nt_store(wp + tid + 256 * i, v[i] * inv);
}

// Row softmax over 1024 bf16 elements, in place (attn, re-read by AV GEMM).
__global__ void softmax1024_bf16(__bf16* __restrict__ data) {
  __shared__ float red[256];
  const size_t row = blockIdx.x;
  __bf16* rp = data + row * 1024;
  const int tid = threadIdx.x;
  float v[4]; float mx = -3.4e38f;
#pragma unroll
  for (int i = 0; i < 4; ++i) { v[i] = (float)rp[tid + 256 * i]; mx = fmaxf(mx, v[i]); }
  red[tid] = mx; __syncthreads();
  for (int s = 128; s > 0; s >>= 1) {
    if (tid < s) red[tid] = fmaxf(red[tid], red[tid + s]);
    __syncthreads();
  }
  mx = red[0]; __syncthreads();
  float sum = 0.f;
#pragma unroll
  for (int i = 0; i < 4; ++i) { v[i] = __expf(v[i] - mx); sum += v[i]; }
  red[tid] = sum; __syncthreads();
  for (int s = 128; s > 0; s >>= 1) {
    if (tid < s) red[tid] += red[tid + s];
    __syncthreads();
  }
  const float inv = 1.0f / red[0];
#pragma unroll
  for (int i = 0; i < 4; ++i) rp[tid + 256 * i] = (__bf16)(v[i] * inv);
}

// ---------------------------------------------------------------------------
// out = attn @ V per head, merged to [b*1024+s][h*64+d] bf16.
// One wave computes 16(q) x 64(d = whole head): 5 loads, 4 WMMAs per k-step.
// attn: [bh][1024][1024] bf16.  Vt: [bh][64][1024] bf16 (k-contig).
// Waves: 64(bh) * 64(qt) = 4096 -> 512 blocks.
// ---------------------------------------------------------------------------
__global__ void av_kernel(const __bf16* __restrict__ attn,
                          const __bf16* __restrict__ Vt,
                          __bf16* __restrict__ omb) {   // [4096][1024]
  const int wave = (blockIdx.x * blockDim.x + threadIdx.x) >> 5;
  const int lane = threadIdx.x & 31;
  const int qt = wave & 63;
  const int bh = wave >> 6;

  const __bf16* Ab = attn + ((size_t)bh * 1024 + qt * 16) * 1024;
  const __bf16* Vb = Vt + (size_t)bh * 64 * 1024;

  v8f acc[4] = {{}, {}, {}, {}};
  for (int k0 = 0; k0 < 1024; k0 += 32) {
    v16bf a = load_frag16(Ab, 1024, k0, lane);
#pragma unroll
    for (int j = 0; j < 4; ++j) {
      v16bf b = load_frag16(Vb + (size_t)(j * 16) * 1024, 1024, k0, lane);
      acc[j] = wmma_bf16(a, b, acc[j]);
    }
  }

  const int sel = lane >> 4;
  const int b = bh >> 4, hd = bh & 15;
#pragma unroll
  for (int j = 0; j < 4; ++j) {
    const int n = hd * 64 + j * 16 + (lane & 15);
#pragma unroll
    for (int v = 0; v < 8; ++v) {
      const int m = b * 1024 + qt * 16 + sel * 8 + v;
      omb[(size_t)m * 1024 + n] = (__bf16)acc[j][v];
    }
  }
}

// ---------------------------------------------------------------------------
// y = h + ALPHA * (out_merge @ o_fwd^T).
// One wave computes 16(M) x 64(N): 5 loads, 4 WMMAs per k-step.
// Waves: 256(mt) * 16(ngroup) = 4096 -> 512 blocks.
// ---------------------------------------------------------------------------
__global__ void out_proj_kernel(const __bf16* __restrict__ omb,   // [4096][1024]
                                const __bf16* __restrict__ Wo,    // [1024][1024]
                                const float*  __restrict__ h,     // [4096][1024]
                                float* __restrict__ y) {
  const int wave = (blockIdx.x * blockDim.x + threadIdx.x) >> 5;
  const int lane = threadIdx.x & 31;
  const int ng = wave & 15;           // 64-wide n group
  const int mt = wave >> 4;           // 0..255
  const int m0 = mt * 16, n0 = ng * 64;

  const __bf16* Ab = omb + (size_t)m0 * 1024;
  const __bf16* Wb = Wo  + (size_t)n0 * 1024;

  v8f acc[4] = {{}, {}, {}, {}};
  for (int k0 = 0; k0 < 1024; k0 += 32) {
    v16bf a = load_frag16(Ab, 1024, k0, lane);
#pragma unroll
    for (int j = 0; j < 4; ++j) {
      v16bf b = load_frag16(Wb + (size_t)(j * 16) * 1024, 1024, k0, lane);
      acc[j] = wmma_bf16(a, b, acc[j]);
    }
  }

  const int sel = lane >> 4;
#pragma unroll
  for (int j = 0; j < 4; ++j) {
    const int n = n0 + j * 16 + (lane & 15);
#pragma unroll
    for (int v = 0; v < 8; ++v) {
      const int m = m0 + sel * 8 + v;
      const size_t idx = (size_t)m * 1024 + n;
      nt_store(y + idx, h[idx] + ALPHA_C * acc[j][v]);
    }
  }
}

// ---------------------------------------------------------------------------
// Host-side orchestration
// ---------------------------------------------------------------------------
extern "C" void kernel_launch(void* const* d_in, const int* in_sizes, int n_in,
                              void* d_out, int out_size, void* d_ws, size_t ws_size,
                              hipStream_t stream) {
  const float* h_f32 = (const float*)d_in[0];
  const float* w_f32[8];
  for (int i = 0; i < 8; ++i) w_f32[i] = (const float*)d_in[1 + i];

  const size_t SZ_H = 4096UL * 1024;      // h / out_merge elements
  const size_t SZ_W = 1024UL * 1024;      // weight elements
  const size_t SZ_P = 64UL * 1024 * 64;   // per-projection [bh][s][d]
  const size_t SZ_E = 64UL * 1024 * 1024; // energy map

  // Workspace carve (bf16): ~200 MB total.
  char* w = (char*)d_ws;
  __bf16* hbf = (__bf16*)w;               w += SZ_H * 2;
  __bf16* wbf[8];
  for (int i = 0; i < 8; ++i) { wbf[i] = (__bf16*)w; w += SZ_W * 2; }
  __bf16* Qbf  = (__bf16*)w;              w += SZ_P * 2;
  __bf16* Kbf  = (__bf16*)w;              w += SZ_P * 2;
  __bf16* Qrbf = (__bf16*)w;              w += SZ_P * 2;
  __bf16* Krbf = (__bf16*)w;              w += SZ_P * 2;
  __bf16* Vtbf = (__bf16*)w;              w += SZ_P * 2;  // [bh][d][s]
  __bf16* ebf  = (__bf16*)w;              w += SZ_E * 2;  // scaled energy -> attn
  __bf16* omb  = (__bf16*)w;              w += SZ_H * 2;  // merged attn output

  // d_out sections (fp32, reference return order)
  float* y_out  = (float*)d_out;
  float* qd_out = y_out + SZ_H;
  float* kd_out = qd_out + SZ_P;
  float* vd_out = kd_out + SZ_P;
  float* ed_out = vd_out + SZ_P;          // SCALING*(e - er)
  float* am_out = ed_out + SZ_E;          // att_minus

  // 1) fp32 -> bf16 conversions (o_fwd_copy unused by the reference).
  cvt_f32_bf16<<<(int)(SZ_H / 256), 256, 0, stream>>>(h_f32, hbf, (int)SZ_H);
  for (int i = 0; i < 7; ++i)
    cvt_f32_bf16<<<(int)(SZ_W / 256), 256, 0, stream>>>(w_f32[i], wbf[i], (int)SZ_W);

  // 2) fused projection pairs (8192 waves each -> 1024 blocks of 8 waves)
  proj_pair_kernel<<<1024, 256, 0, stream>>>(hbf, wbf[0], wbf[4], qd_out,
                                             Qbf, Qrbf, nullptr);
  proj_pair_kernel<<<1024, 256, 0, stream>>>(hbf, wbf[1], wbf[5], kd_out,
                                             Kbf, Krbf, nullptr);
  proj_pair_kernel<<<1024, 256, 0, stream>>>(hbf, wbf[2], wbf[6], vd_out,
                                             nullptr, nullptr, Vtbf);

  // 3) energies + scaled diff (65536 waves -> 8192 blocks)
  energy_kernel<<<8192, 256, 0, stream>>>(Qbf, Kbf, Qrbf, Krbf, ed_out, ebf);

  // 4) softmaxes: att_minus (fp32) and attn (bf16, in place)
  softmax1024_f32<<<65536, 256, 0, stream>>>(ed_out, am_out);
  softmax1024_bf16<<<65536, 256, 0, stream>>>(ebf);

  // 5) attn @ V -> merged bf16 (4096 waves -> 512 blocks)
  av_kernel<<<512, 256, 0, stream>>>(ebf, Vtbf, omb);

  // 6) y = h + ALPHA * (out_merge @ o_fwd^T) (4096 waves -> 512 blocks)
  out_proj_kernel<<<512, 256, 0, stream>>>(omb, wbf[3], h_f32, y_out);
}
